// InteractionBlock_23003844837590
// MI455X (gfx1250) — compile-verified, matrix operand order checked by reading, not run on hardware
//
#include <hip/hip_runtime.h>

// ---- problem constants ----
#define NB   16
#define MM   20
#define CC   2304
#define DIM  1024
#define SS   196            // T*H*W = 4*7*7
#define LL   4
#define PPG  5              // M / L
#define NKEY 3920           // M * S
#define MSP  3968           // NKEY padded to 31*128 (multiple of 32 and 128)
#define QKSCALE 0.03125f    // DI^-0.5
#define LN_EPS 1e-5f

typedef __attribute__((ext_vector_type(8)))  float        v8f;
typedef __attribute__((ext_vector_type(16))) __bf16       v16bf;
typedef __attribute__((ext_vector_type(4)))  unsigned int u32x4;
typedef __attribute__((ext_vector_type(2)))  unsigned int u32x2;
typedef __attribute__((ext_vector_type(4)))  float        f32x4;

union Frag { u32x4 u[2]; v16bf v; };

__device__ __forceinline__ unsigned short f2bf(float f) {
  unsigned int u = __float_as_uint(f);
  u += 0x7FFFu + ((u >> 16) & 1u);   // round-to-nearest-even
  return (unsigned short)(u >> 16);
}

#define LDS_TILE (128 * 40)   // one buffer: 128 rows x (32+8 pad) halves

// =====================================================================
// Shared WMMA GEMM core.  Block = 256 threads (8 waves, 4 row-groups x
// 2 col-groups).  Block tile: 128 rows x 128 cols, K advanced 32/step.
// Per wave per step: 2 A-frags x 4 B-frags -> 8 v_wmma_f32_16x16x32_bf16,
// B-fragments loaded one at a time to keep VGPR pressure low (no spills).
// Double-buffered LDS, software-pipelined global->LDS staging; fp32
// activations are converted to packed bf16 at load time so only 8 VGPRs
// of staged A data stay live across the WMMA burst.
// APATH 0: A fp32 "activation" layout, element(row,k) = A32[k*aStride+row]
// APATH 1: A bf16 row-major, element(row,k) = A16[row*aStride+k]
// B bf16 column-major-for-WMMA: element(col,k) = Bp[col*bStride+k]
// =====================================================================
template <int APATH>
__device__ __forceinline__ void gemm_core(const float* __restrict__ A32,
                                          const unsigned short* __restrict__ A16,
                                          int aStride,
                                          const unsigned short* __restrict__ Bp,
                                          int bStride, int K,
                                          unsigned short* ldsA,
                                          unsigned short* ldsB,
                                          v8f acc[8]) {
  const int tid  = threadIdx.x;
  const int lane = tid & 31;
  const int wv   = tid >> 5;
  const int rg   = wv >> 1;                 // row group 0..3
  const int cg   = wv & 1;                  // col group 0..1
  const int run0 = (lane & 16) ? 8 : 0;     // K-run start per ISA A/B layout
  const int arow0 = rg * 32 + (lane & 15);

  // staging thread mapping
  const int kq   = (tid >> 5) * 4;          // APATH0: k quad 0,4,...,28
  const int r4   = (tid & 31) * 4;          // APATH0: row quad 0..124
  int r4c = r4;
  if (APATH == 0) { const int lim = aStride - 4; if (r4c > lim) r4c = lim; }
  const int srow = tid >> 2;                // APATH1 A & B: seg row 0..63
  const int part = (tid & 3) * 8;           // seg k-offset {0,8,16,24}

  u32x2 pa0, pa1, pa2, pa3;                 // APATH0 staged (packed bf16)
  u32x4 ra0, ra1;                           // APATH1 staged registers
  u32x4 rb0, rb1;                           // B staged registers

  auto loadStage = [&](int ks) {
    if (APATH == 0) {
      f32x4 fa0 = *(const f32x4*)(A32 + (long)(ks + kq + 0) * aStride + r4c);
      f32x4 fa1 = *(const f32x4*)(A32 + (long)(ks + kq + 1) * aStride + r4c);
      f32x4 fa2 = *(const f32x4*)(A32 + (long)(ks + kq + 2) * aStride + r4c);
      f32x4 fa3 = *(const f32x4*)(A32 + (long)(ks + kq + 3) * aStride + r4c);
      // transpose 4x4 and pack to bf16 immediately (short fp32 lifetimes)
#define PACKROW(DST, FLD)                                                      \
      { DST.x = (unsigned)f2bf(fa0.FLD) | ((unsigned)f2bf(fa1.FLD) << 16);     \
        DST.y = (unsigned)f2bf(fa2.FLD) | ((unsigned)f2bf(fa3.FLD) << 16); }
      PACKROW(pa0, x) PACKROW(pa1, y) PACKROW(pa2, z) PACKROW(pa3, w)
#undef PACKROW
    } else {
      ra0 = *(const u32x4*)(A16 + (long)srow * aStride + ks + part);
      ra1 = *(const u32x4*)(A16 + (long)(srow + 64) * aStride + ks + part);
    }
    rb0 = *(const u32x4*)(Bp + (long)srow * bStride + ks + part);
    rb1 = *(const u32x4*)(Bp + (long)(srow + 64) * bStride + ks + part);
  };

  auto storeStage = [&](unsigned short* dA, unsigned short* dB) {
    if (APATH == 0) {
      *(u32x2*)&dA[(r4 + 0) * 40 + kq] = pa0;
      *(u32x2*)&dA[(r4 + 1) * 40 + kq] = pa1;
      *(u32x2*)&dA[(r4 + 2) * 40 + kq] = pa2;
      *(u32x2*)&dA[(r4 + 3) * 40 + kq] = pa3;
    } else {
      *(u32x4*)&dA[srow * 40 + part]        = ra0;
      *(u32x4*)&dA[(srow + 64) * 40 + part] = ra1;
    }
    *(u32x4*)&dB[srow * 40 + part]        = rb0;
    *(u32x4*)&dB[(srow + 64) * 40 + part] = rb1;
  };

  const int nSteps = K >> 5;
  loadStage(0);
  storeStage(ldsA, ldsB);

#pragma unroll 2
  for (int ksi = 0; ksi < nSteps; ++ksi) {
    const int cur = ksi & 1;
    unsigned short* cA = ldsA + cur * LDS_TILE;
    unsigned short* cB = ldsB + cur * LDS_TILE;
    unsigned short* nA = ldsA + (cur ^ 1) * LDS_TILE;
    unsigned short* nB = ldsB + (cur ^ 1) * LDS_TILE;

    if (ksi + 1 < nSteps) loadStage((ksi + 1) << 5);  // overlap with WMMAs
    if (ksi + 2 < nSteps) {                           // L2 prefetch 2 ahead
      const int ks2 = (ksi + 2) << 5;
      __builtin_prefetch(Bp + (long)srow * bStride + ks2 + part, 0, 1);
      if (APATH == 0)
        __builtin_prefetch(A32 + (long)(ks2 + kq) * aStride + r4c, 0, 1);
      else
        __builtin_prefetch(A16 + (long)srow * aStride + ks2 + part, 0, 1);
    }
    __syncthreads();   // buffer `cur` ready (stores from prev iteration done)

    Frag a0, a1;
    a0.u[0] = *(const u32x4*)&cA[arow0 * 40 + run0];
    a0.u[1] = *(const u32x4*)&cA[arow0 * 40 + run0 + 16];
    a1.u[0] = *(const u32x4*)&cA[(arow0 + 16) * 40 + run0];
    a1.u[1] = *(const u32x4*)&cA[(arow0 + 16) * 40 + run0 + 16];
    const int bc = cg * 64 + (lane & 15);
#pragma unroll
    for (int ct = 0; ct < 4; ++ct) {
      Frag b;
      b.u[0] = *(const u32x4*)&cB[(bc + ct * 16) * 40 + run0];
      b.u[1] = *(const u32x4*)&cB[(bc + ct * 16) * 40 + run0 + 16];
      acc[ct]     = __builtin_amdgcn_wmma_f32_16x16x32_bf16(
          false, a0.v, false, b.v, (short)0, acc[ct], false, false);
      acc[4 + ct] = __builtin_amdgcn_wmma_f32_16x16x32_bf16(
          false, a1.v, false, b.v, (short)0, acc[4 + ct], false, false);
    }

    if (ksi + 1 < nSteps) storeStage(nA, nB);  // fill next buffer
  }
}

#define GEMM_PROLOGUE()                                                       \
  __shared__ unsigned short ldsA[2 * LDS_TILE];                               \
  __shared__ unsigned short ldsB[2 * LDS_TILE];                               \
  const int z = blockIdx.z;                                                   \
  const int rowBase = blockIdx.y * 128;                                       \
  const int colBase = blockIdx.x * 128;                                       \
  v8f acc[8];                                                                 \
  {                                                                           \
    v8f zr = {0.f, 0.f, 0.f, 0.f, 0.f, 0.f, 0.f, 0.f};                        \
    for (int i = 0; i < 8; ++i) acc[i] = zr;                                  \
  }                                                                           \
  const int lane = threadIdx.x & 31;                                          \
  const int wv = threadIdx.x >> 5;                                            \
  const int rg = wv >> 1;                                                     \
  const int cg = wv & 1;                                                      \
  const int mrow = 8 * (lane >> 4);

// ---------------------------------------------------------------------
// QKV projection: q/k/v = act x W^T + bias(+tp).  mode 0=q, 1=k, 2=v(T)
// ---------------------------------------------------------------------
__global__ void __launch_bounds__(256, 1)
k_proj(const float* __restrict__ act,
       const unsigned short* __restrict__ W,
       const float* __restrict__ bias, int selL,
       unsigned short* __restrict__ out, int mode) {
  GEMM_PROLOGUE();
  const float* A32 = act + (long)z * CC * SS + rowBase;
  const unsigned short* Bp = W + (long)colBase * CC;

  const float* bvec = bias;
  long obase; int rst, cst;
  if (mode == 0) {
    obase = (long)z * SS * DIM; rst = DIM; cst = 1;
  } else {
    const int n = z / MM, m = z % MM, l = m / PPG;
    if (selL) bvec = bias + l * DIM;
    if (mode == 1) { obase = ((long)n * MSP + (long)m * SS) * DIM; rst = DIM; cst = 1; }
    else           { obase = (long)n * DIM * MSP + (long)m * SS;   rst = 1;   cst = MSP; }
  }

  gemm_core<0>(A32, nullptr, SS, Bp, CC, CC, ldsA, ldsB, acc);

#pragma unroll
  for (int ar = 0; ar < 2; ++ar) {
    const int rb2 = rowBase + rg * 32 + ar * 16 + mrow;
#pragma unroll
    for (int ct = 0; ct < 4; ++ct) {
      const int gcol = colBase + cg * 64 + ct * 16 + (lane & 15);
      const float bs = bvec[gcol];
      const v8f a = acc[ar * 4 + ct];
#pragma unroll
      for (int i = 0; i < 8; ++i) {
        const int grow = rb2 + i;
        if (grow < SS)
          out[obase + (long)grow * rst + (long)gcol * cst] = f2bf(a[i] + bs);
      }
    }
  }
}

// ---------------------------------------------------------------------
// scores = (q x k^T) * SCALE   (per n: 196 x 3968, K = 1024)
// ---------------------------------------------------------------------
__global__ void __launch_bounds__(256, 1)
k_scores(const unsigned short* __restrict__ q,
         const unsigned short* __restrict__ kbuf,
         float* __restrict__ sc) {
  GEMM_PROLOGUE();
  const unsigned short* A16 = q + (long)z * SS * DIM + (long)rowBase * DIM;
  const unsigned short* Bp  = kbuf + ((long)z * MSP + colBase) * DIM;
  gemm_core<1>(nullptr, A16, DIM, Bp, DIM, DIM, ldsA, ldsB, acc);
#pragma unroll
  for (int ar = 0; ar < 2; ++ar) {
    const int rb2 = rowBase + rg * 32 + ar * 16 + mrow;
#pragma unroll
    for (int ct = 0; ct < 4; ++ct) {
      const int gcol = colBase + cg * 64 + ct * 16 + (lane & 15);
      const v8f a = acc[ar * 4 + ct];
#pragma unroll
      for (int i = 0; i < 8; ++i) {
        const int grow = rb2 + i;
        if (grow < SS)
          sc[(long)z * SS * MSP + (long)grow * MSP + gcol] = a[i] * QKSCALE;
      }
    }
  }
}

// ---------------------------------------------------------------------
// o = attn x v   (per n: 196 x 1024, K = 3968; v stored [d][key])
// ---------------------------------------------------------------------
__global__ void __launch_bounds__(256, 1)
k_av(const unsigned short* __restrict__ attn,
     const unsigned short* __restrict__ vT,
     float* __restrict__ o) {
  GEMM_PROLOGUE();
  const unsigned short* A16 = attn + (long)z * SS * MSP + (long)rowBase * MSP;
  const unsigned short* Bp  = vT + (long)z * DIM * MSP + (long)colBase * MSP;
  gemm_core<1>(nullptr, A16, MSP, Bp, MSP, MSP, ldsA, ldsB, acc);
#pragma unroll
  for (int ar = 0; ar < 2; ++ar) {
    const int rb2 = rowBase + rg * 32 + ar * 16 + mrow;
#pragma unroll
    for (int ct = 0; ct < 4; ++ct) {
      const int gcol = colBase + cg * 64 + ct * 16 + (lane & 15);
      const v8f a = acc[ar * 4 + ct];
#pragma unroll
      for (int i = 0; i < 8; ++i) {
        const int grow = rb2 + i;
        if (grow < SS)
          o[(long)z * SS * DIM + (long)grow * DIM + gcol] = a[i];
      }
    }
  }
}

// ---------------------------------------------------------------------
// out[n][c][pos] = oln x Wo^T + bo + person  (per n: 196 x 2304, K=1024)
// ---------------------------------------------------------------------
__global__ void __launch_bounds__(256, 1)
k_out(const unsigned short* __restrict__ oln,
      const unsigned short* __restrict__ wo,
      const float* __restrict__ bo,
      const float* __restrict__ person,
      float* __restrict__ outp) {
  GEMM_PROLOGUE();
  const unsigned short* A16 = oln + (long)z * SS * DIM + (long)rowBase * DIM;
  const unsigned short* Bp  = wo + (long)colBase * DIM;
  gemm_core<1>(nullptr, A16, DIM, Bp, DIM, DIM, ldsA, ldsB, acc);
#pragma unroll
  for (int ar = 0; ar < 2; ++ar) {
    const int rb2 = rowBase + rg * 32 + ar * 16 + mrow;
#pragma unroll
    for (int ct = 0; ct < 4; ++ct) {
      const int gcol = colBase + cg * 64 + ct * 16 + (lane & 15);
      const float bs = bo[gcol];
      const v8f a = acc[ar * 4 + ct];
#pragma unroll
      for (int i = 0; i < 8; ++i) {
        const int grow = rb2 + i;
        if (grow < SS) {
          const long idx = (long)z * CC * SS + (long)gcol * SS + grow;
          outp[idx] = a[i] + bs + person[idx];
        }
      }
    }
  }
}

// ---------------------------------------------------------------------
// helpers: weight conversion, bias folding, softmax, layernorm
// ---------------------------------------------------------------------
__global__ void k_cvtw(const float* __restrict__ wq, const float* __restrict__ wk,
                       const float* __restrict__ wv, const float* __restrict__ wo,
                       unsigned short* __restrict__ dq, unsigned short* __restrict__ dk,
                       unsigned short* __restrict__ dv, unsigned short* __restrict__ dwo) {
  const long i = (long)blockIdx.x * 256 + threadIdx.x;
  const long per = (long)DIM * CC;
  if (i >= 4 * per) return;
  const long t = i / per, off = i % per;
  const float* s = (t == 0) ? wq : (t == 1) ? wk : (t == 2) ? wv : wo;
  unsigned short* d = (t == 0) ? dq : (t == 1) ? dk : (t == 2) ? dv : dwo;
  d[off] = f2bf(s[off]);
}

__global__ void k_bias(const float* __restrict__ bk, const float* __restrict__ tpk,
                       const float* __restrict__ bv_, const float* __restrict__ tpv,
                       float* __restrict__ bK, float* __restrict__ bV) {
  const int i = blockIdx.x * 256 + threadIdx.x;   // < 2*L*DIM
  if (i < LL * DIM) bK[i] = bk[i & (DIM - 1)] + tpk[i];
  else { const int j = i - LL * DIM; bV[j] = bv_[j & (DIM - 1)] + tpv[j]; }
}

__global__ void k_softmax(float* __restrict__ sc, unsigned short* __restrict__ attn) {
  __shared__ float red[256];
  const int b = blockIdx.x;              // n*196 rows
  const int n = b / SS, r = b % SS;
  float* row = sc + (long)n * SS * MSP + (long)r * MSP;
  unsigned short* arow = attn + (long)n * SS * MSP + (long)r * MSP;
  const int tid = threadIdx.x;
  float mx = -3.0e38f;
  for (int c = tid; c < NKEY; c += 256) mx = fmaxf(mx, row[c]);
  red[tid] = mx; __syncthreads();
  for (int s = 128; s > 0; s >>= 1) {
    if (tid < s) red[tid] = fmaxf(red[tid], red[tid + s]);
    __syncthreads();
  }
  mx = red[0]; __syncthreads();
  float sum = 0.f;
  for (int c = tid; c < NKEY; c += 256) {
    const float e = __expf(row[c] - mx);
    row[c] = e; sum += e;
  }
  red[tid] = sum; __syncthreads();
  for (int s = 128; s > 0; s >>= 1) {
    if (tid < s) red[tid] += red[tid + s];
    __syncthreads();
  }
  const float inv = 1.0f / red[0];
  for (int c = tid; c < MSP; c += 256)
    arow[c] = (c < NKEY) ? f2bf(row[c] * inv) : (unsigned short)0;  // zero the pad
}

__global__ void k_ln(const float* __restrict__ o, unsigned short* __restrict__ oln) {
  __shared__ float r1[256], r2[256];
  const int n = blockIdx.x, tid = threadIdx.x;
  const float* p = o + (long)n * SS * DIM;
  float s = 0.f, s2 = 0.f;
  for (int i = tid; i < SS * DIM; i += 256) { const float v = p[i]; s += v; s2 += v * v; }
  r1[tid] = s; r2[tid] = s2; __syncthreads();
  for (int st = 128; st > 0; st >>= 1) {
    if (tid < st) { r1[tid] += r1[tid + st]; r2[tid] += r2[tid + st]; }
    __syncthreads();
  }
  const float inv_n = 1.0f / (float)(SS * DIM);
  const float mu = r1[0] * inv_n;
  const float var = r2[0] * inv_n - mu * mu;
  const float rs = rsqrtf(var + LN_EPS);
  unsigned short* q = oln + (long)n * SS * DIM;
  for (int i = tid; i < SS * DIM; i += 256) {
    const float v = (p[i] - mu) * rs;
    q[i] = f2bf(v > 0.f ? v : 0.f);
  }
}

// ---------------------------------------------------------------------
extern "C" void kernel_launch(void* const* d_in, const int* in_sizes, int n_in,
                              void* d_out, int out_size, void* d_ws, size_t ws_size,
                              hipStream_t stream) {
  (void)in_sizes; (void)n_in; (void)out_size; (void)ws_size;
  const float* person = (const float*)d_in[0];
  const float* others = (const float*)d_in[1];
  const float* Wq = (const float*)d_in[2];
  const float* bq = (const float*)d_in[3];
  const float* Wk = (const float*)d_in[4];
  const float* bk = (const float*)d_in[5];
  const float* Wv = (const float*)d_in[6];
  const float* bv = (const float*)d_in[7];
  const float* Wo = (const float*)d_in[8];
  const float* bo = (const float*)d_in[9];
  const float* tpk = (const float*)d_in[10];
  const float* tpv = (const float*)d_in[11];
  float* outp = (float*)d_out;

  // workspace carve-up (ordered so A-tile row overruns stay inside d_ws)
  char* w = (char*)d_ws;
  auto carve = [&](size_t bytes) {
    char* p = w; w += (bytes + 255) & ~(size_t)255; return p;
  };
  unsigned short* qb   = (unsigned short*)carve((size_t)NB * SS * DIM * 2);
  unsigned short* attn = (unsigned short*)carve((size_t)NB * SS * MSP * 2);
  unsigned short* oln  = (unsigned short*)carve((size_t)NB * SS * DIM * 2);
  unsigned short* wqb  = (unsigned short*)carve((size_t)DIM * CC * 2);
  unsigned short* wkb  = (unsigned short*)carve((size_t)DIM * CC * 2);
  unsigned short* wvb  = (unsigned short*)carve((size_t)DIM * CC * 2);
  unsigned short* wob  = (unsigned short*)carve((size_t)CC * DIM * 2);
  float*          bK   = (float*)carve((size_t)LL * DIM * 4);
  float*          bV   = (float*)carve((size_t)LL * DIM * 4);
  unsigned short* kb   = (unsigned short*)carve((size_t)NB * MSP * DIM * 2);
  unsigned short* vT   = (unsigned short*)carve((size_t)NB * DIM * MSP * 2);
  float*          sc   = (float*)carve((size_t)NB * SS * MSP * 4);
  float*          ob   = (float*)carve((size_t)NB * SS * DIM * 4);

  const dim3 blk(256);
  k_cvtw<<<dim3(36864), blk, 0, stream>>>(Wq, Wk, Wv, Wo, wqb, wkb, wvb, wob);
  k_bias<<<dim3(32), blk, 0, stream>>>(bk, tpk, bv, tpv, bK, bV);
  k_proj<<<dim3(8, 2, 16),  blk, 0, stream>>>(person, wqb, bq, 0, qb, 0);
  k_proj<<<dim3(8, 2, 320), blk, 0, stream>>>(others, wkb, bK, 1, kb, 1);
  k_proj<<<dim3(8, 2, 320), blk, 0, stream>>>(others, wvb, bV, 1, vT, 2);
  k_scores<<<dim3(31, 2, 16), blk, 0, stream>>>(qb, kb, sc);
  k_softmax<<<dim3(NB * SS), blk, 0, stream>>>(sc, attn);
  k_av<<<dim3(8, 2, 16), blk, 0, stream>>>(attn, vT, ob);
  k_ln<<<dim3(16), blk, 0, stream>>>(ob, oln);
  k_out<<<dim3(18, 2, 16), blk, 0, stream>>>(oln, wob, bo, person, outp);
}
